// OverlappingWindow_58437325030107
// MI455X (gfx1250) — matrix operand order repaired and checked by reading, not compiled, not run on hardware
//
#include <hip/hip_runtime.h>
#include <hip/hip_bf16.h>

typedef __bf16 bf16;
typedef __attribute__((ext_vector_type(16))) __bf16 v16bf;
typedef __attribute__((ext_vector_type(8)))  __bf16 v8bf;
typedef __attribute__((ext_vector_type(8)))  float  v8f;
typedef __attribute__((ext_vector_type(4)))  unsigned int v4u;
typedef __attribute__((ext_vector_type(8)))  int v8i;
typedef __attribute__((ext_vector_type(4)))  int v4i;

#define BATCH 8192

#if defined(__has_builtin)
#if __has_builtin(__builtin_amdgcn_tensor_load_to_lds) && __has_builtin(__builtin_amdgcn_s_wait_tensorcnt)
#define USE_TDM 1
#endif
#endif
#ifndef USE_TDM
#define USE_TDM 0
#endif

// ---------------------------------------------------------------------------
// Weight packing
// ---------------------------------------------------------------------------
__global__ __launch_bounds__(256) void pack_mat(const float* __restrict__ src,
                                                bf16* __restrict__ dst,
                                                int M, int K, int Kpad, int Mpad) {
  long idx = (long)blockIdx.x * 256 + threadIdx.x;
  long total = (long)Mpad * Kpad;
  if (idx >= total) return;
  int m = (int)(idx / Kpad);
  int k = (int)(idx - (long)m * Kpad);
  float v = (m < M && k < K) ? src[(long)m * K + k] : 0.0f;
  dst[idx] = (bf16)v;
}

// Conv weights OIHW (68,68,3,3) -> tap-major bf16 [80 x 736], col = t*80 + ci.
__global__ __launch_bounds__(256) void pack_conv(const float* __restrict__ src,
                                                 bf16* __restrict__ dst) {
  int idx = blockIdx.x * 256 + threadIdx.x;
  if (idx >= 80 * 736) return;
  int m = idx / 736; int k = idx - m * 736;
  int t = k / 80;    int ci = k - t * 80;
  float v = 0.0f;
  if (m < 68 && t < 9 && ci < 68) v = src[((size_t)m * 68 + ci) * 9 + t];
  dst[idx] = (bf16)v;
}

__global__ __launch_bounds__(256) void pack_vecf(const float* __restrict__ src,
                                                 float* __restrict__ dst, int M, int Mpad) {
  int i = blockIdx.x * 256 + threadIdx.x;
  if (i < Mpad) dst[i] = (i < M) ? src[i] : 0.0f;
}

__global__ __launch_bounds__(256) void zero_pad_rows(bf16* __restrict__ Bm,
                                                     int Kreal, int KlPad, int NB) {
  int pad = KlPad - Kreal;
  long total = (long)NB * pad;
  long i = (long)blockIdx.x * 256 + threadIdx.x;
  if (i >= total) return;
  long n = i / pad;
  int  r = (int)(i - n * pad);
  Bm[n * KlPad + Kreal + r] = (bf16)0.0f;
}

// ---------------------------------------------------------------------------
// Pre-activated window buffer WITH 1-pixel zero halo:
// Z[nb][(qh)*(ws+2)+(qw)][80ch], interior at qh,qw in [1..ws]; halo = 0.
// ---------------------------------------------------------------------------
__global__ __launch_bounds__(256) void act_win(const float* __restrict__ x_old,
                                               const float* __restrict__ cellp,
                                               const float* __restrict__ iscp,
                                               bf16* __restrict__ Z, int ws) {
  int wsp = ws + 2, wsp2 = wsp * wsp;
  long idx = (long)blockIdx.x * 256 + threadIdx.x;
  long total = (long)4 * BATCH * wsp2 * 80;
  if (idx >= total) return;
  int ci = (int)(idx % 80);
  long r0 = idx / 80;
  int q  = (int)(r0 % wsp2);
  int nb = (int)(r0 / wsp2);
  int corner = nb >> 13, b = nb & (BATCH - 1);
  int qh = q / wsp, qw = q - qh * wsp;
  bf16 outv = (bf16)0.0f;
  if (ci < 68 && qh >= 1 && qh <= ws && qw >= 1 && qw <= ws) {
    int gh = ((corner & 2) ? (6 - ws) : 0) + qh - 1;
    int gw = ((corner & 1) ? (6 - ws) : 0) + qw - 1;
    float v;
    if (ci < 64)      v = x_old[((size_t)b * 64 + ci) * 36 + gh * 6 + gw];
    else if (ci < 66) v = cellp[b * 2 + (ci - 64)];
    else              v = iscp[((size_t)b * 2 + (ci - 66)) * 36 + gh * 6 + gw];
    outv = (bf16)(__sinf(v) + __cosf(v));
  }
  Z[idx] = outv;
}

// ---------------------------------------------------------------------------
// Fused WMMA GEMM (NT=2 n-tiles per wave, double-buffered TDM A staging).
//  MODE 0: conv1 — implicit im2col from halo'd Zbuf (branchless v16bf load);
//                  epilogue writes sin+cos(relu(y)) into halo'd OutB
//  MODE 1: conv3 — same B path; epilogue writes relu(y) -> OutB[n][k]
//  MODE 2: stage linear — B = plain bf16 [n][KlPad]; epilogue scatters 6x6 x
//  MODE 3: final linear — B gathered from corner windows; epilogue fuses xc*
// ---------------------------------------------------------------------------
template<int MODE, int MT>
__global__ __launch_bounds__(128) void gemm_k(
    const bf16*  __restrict__ Wp,   const float* __restrict__ biasp,
    const bf16*  __restrict__ Zbuf,
    bf16*  __restrict__ OutB,
    float* __restrict__ x_new,      float* __restrict__ outp,
    const float* __restrict__ xfin,
    int Ktiles, int Kw, int ws, int KlPad)
{
  __shared__ __align__(16) bf16 sA[2][MT * 16 * 32];

  const int tid   = threadIdx.x;
  const int wave  = tid >> 5;
  const int lane  = tid & 31;
  const int lrow  = lane & 15;
  const int lhalf = lane >> 4;
  const int n0    = blockIdx.x * 128 + wave * 16;   // second n-tile at +64
  const int mbase = blockIdx.y * MT * 16;
  const int ws2   = ws * ws;
  const int wsp   = ws + 2, wsp2 = wsp * wsp;

  int ncol[2], nb[2], p[2], ph[2], pw[2], corner[2], bidx[2];
  #pragma unroll
  for (int j = 0; j < 2; ++j) {
    ncol[j] = n0 + 64 * j + lrow;
    if (MODE == 0 || MODE == 1) {
      nb[j] = ncol[j] / ws2; p[j] = ncol[j] - nb[j] * ws2;
      ph[j] = p[j] / ws;     pw[j] = p[j] - ph[j] * ws;
      corner[j] = 0; bidx[j] = 0;
    } else {
      corner[j] = ncol[j] >> 13; bidx[j] = ncol[j] & (BATCH - 1);
      nb[j] = 0; p[j] = 0; ph[j] = 0; pw[j] = 0;
    }
  }

  const v8f zinit = {0.f, 0.f, 0.f, 0.f, 0.f, 0.f, 0.f, 0.f};
  v8f acc[MT][2];
  #pragma unroll
  for (int mt = 0; mt < MT; ++mt) { acc[mt][0] = zinit; acc[mt][1] = zinit; }

#if USE_TDM
  auto issueA = [&](int kt) {
    unsigned long long ga =
        (unsigned long long)(size_t)(Wp + (size_t)mbase * Kw + (size_t)kt * 32);
    unsigned lds = (unsigned)(size_t)(void*)(&sA[kt & 1][0]);
    v4u g0;
    g0[0] = 1u;                                        // count=1 user descriptor
    g0[1] = lds;                                       // lds_addr
    g0[2] = (unsigned)(ga & 0xFFFFFFFFu);              // global_addr[31:0]
    g0[3] = (unsigned)((ga >> 32) & 0x01FFFFFFu) | (2u << 30);  // addr[56:32]|type=2
    v8i g1;
    g1[0] = (int)(1u << 16);                                     // data_size = 2B
    g1[1] = (int)(((unsigned)Kw & 0xFFFFu) << 16);               // tensor_dim0 lo
    g1[2] = (int)(((unsigned)Kw >> 16) | ((unsigned)(MT * 16) << 16));
    g1[3] = (int)(32u << 16);                                    // tile_dim0 = 32
    g1[4] = (int)(MT * 16);                                      // tile_dim1
    g1[5] = (int)Kw;                                             // dim0_stride lo
    g1[6] = 0;
    g1[7] = 0;
    v4i z4 = {0, 0, 0, 0};
#if __clang_major__ >= 23
    v8i z8 = {0, 0, 0, 0, 0, 0, 0, 0};
    __builtin_amdgcn_tensor_load_to_lds(g0, g1, z4, z4, z8, 0);
#else
    __builtin_amdgcn_tensor_load_to_lds(g0, g1, z4, z4, 0);
#endif
  };
  if (wave == 0) issueA(0);
#endif

  for (int kt = 0; kt < Ktiles; ++kt) {
    const int cur = kt & 1;
#if USE_TDM
    // wait for tile kt, then (post-barrier) prefetch kt+1 while computing kt
    if (wave == 0) __builtin_amdgcn_s_wait_tensorcnt(0);
    __syncthreads();
    if (wave == 0 && (kt + 1) < Ktiles) issueA(kt + 1);
#else
    __syncthreads();   // previous readers done
    {
      const int chunks = MT * 16 * 4;
      for (int c = tid; c < chunks; c += 128) {
        int row = c >> 2; int seg = c & 3;
        uint4 v = *(const uint4*)(Wp + (size_t)(mbase + row) * Kw + kt * 32 + seg * 8);
        *(uint4*)&sA[cur][row * 32 + seg * 8] = v;
      }
    }
    __syncthreads();
#endif

    // ---- B fragments: 16 contiguous K values per lane, per n-tile ---------
    const int kbase = kt * 32 + lhalf * 16;
    v16bf bfr[2];
    if (MODE == 2) {
      #pragma unroll
      for (int j = 0; j < 2; ++j)
        bfr[j] = *(const v16bf*)(Zbuf + (size_t)ncol[j] * KlPad + kbase);
    } else if (MODE == 0 || MODE == 1) {
      // tap-major k = t*80 + ci ; halo'd buffer -> branchless load
      int t   = kbase / 80;
      int ci0 = kbase - t * 80;
      int dh = t / 3, dw = t - dh * 3;      // window offset (0..2) = tap + 1 - 1
      #pragma unroll
      for (int j = 0; j < 2; ++j) {
        int pp = (ph[j] + dh) * wsp + (pw[j] + dw);   // halo shifts by +1-1
        if (t >= 9) pp = 0;                            // K pad -> zero halo pixel
        bfr[j] = *(const v16bf*)(Zbuf + ((size_t)nb[j] * wsp2 + pp) * 80 + ci0);
      }
    } else { // MODE == 3
      #pragma unroll
      for (int j = 0; j < 2; ++j) {
        #pragma unroll
        for (int jj = 0; jj < 16; ++jj) {
          int k = kbase + jj;
          int ch = k / 9; int t = k - ch * 9;
          int r = t / 3; int cc = t - r * 3;
          int gh = (corner[j] >> 1) * 3 + r;
          int gw = (corner[j] & 1) * 3 + cc;
          bfr[j][jj] = (bf16)xfin[(((size_t)bidx[j] * 64 + ch) * 6 + gh) * 6 + gw];
        }
      }
    }

    // ---- MT x 2 WMMA accumulations ----------------------------------------
    #pragma unroll
    for (int mt = 0; mt < MT; ++mt) {
      const bf16* ap = &sA[cur][(mt * 16 + lrow) * 32 + lhalf * 16];
      v8bf alo = *(const v8bf*)ap;
      v8bf ahi = *(const v8bf*)(ap + 8);
      union { struct { v8bf lo, hi; } s; v16bf v; } u;
      u.s.lo = alo; u.s.hi = ahi;
      acc[mt][0] = __builtin_amdgcn_wmma_f32_16x16x32_bf16(
          false, u.v, false, bfr[0], (short)0, acc[mt][0], false, false);
      acc[mt][1] = __builtin_amdgcn_wmma_f32_16x16x32_bf16(
          false, u.v, false, bfr[1], (short)0, acc[mt][1], false, false);
    }
#if !USE_TDM
#else
#endif
    __syncthreads();
  }

  // ---- epilogue ----
  #pragma unroll
  for (int j = 0; j < 2; ++j) {
    #pragma unroll
    for (int mt = 0; mt < MT; ++mt) {
      #pragma unroll
      for (int r = 0; r < 8; ++r) {
        int mloc = mt * 16 + lhalf * 8 + r;
        int m = mbase + mloc;
        float vacc = acc[mt][j][r];
        if (MODE == 0) {
          float y = fmaxf(vacc + biasp[m], 0.0f);
          OutB[((size_t)nb[j] * wsp2 + (ph[j] + 1) * wsp + (pw[j] + 1)) * 80 + m] =
              (bf16)(__sinf(y) + __cosf(y));
        } else if (MODE == 1) {
          if (mloc < 68) {
            float y = fmaxf(vacc + biasp[m], 0.0f);
            OutB[(size_t)nb[j] * KlPad + m * ws2 + p[j]] = (bf16)y;
          }
        } else if (MODE == 2) {
          float y = fmaxf(vacc + biasp[m], 0.0f);
          int ch = m / 9; int t = m - ch * 9; int rr = t / 3; int cc = t - rr * 3;
          int gh = (corner[j] >> 1) * 3 + rr;
          int gw = (corner[j] & 1) * 3 + cc;
          x_new[(((size_t)bidx[j] * 64 + ch) * 6 + gh) * 6 + gw] = y;
        } else {
          float y = vacc + biasp[m];
          long q = (long)ncol[j] * 64 + m;
          int b2  = (int)(q >> 8);
          int ch2 = (int)((q >> 2) & 63);
          int ii  = (int)((q >> 1) & 1);
          int jj  = (int)(q & 1);
          float xc = xfin[(((size_t)b2 * 64 + ch2) * 6 + (2 + ii)) * 6 + (2 + jj)];
          outp[q] = y * xc;
        }
      }
    }
  }
}

// ---------------------------------------------------------------------------
extern "C" void kernel_launch(void* const* d_in, const int* in_sizes, int n_in,
                              void* d_out, int out_size, void* d_ws, size_t ws_size,
                              hipStream_t stream) {
  (void)in_sizes; (void)n_in; (void)out_size; (void)ws_size;
  const float* x_in  = (const float*)d_in[0];
  const float* cellp = (const float*)d_in[1];
  const float* iscp  = (const float*)d_in[2];
  const float* w1  = (const float*)d_in[3];
  const float* b1  = (const float*)d_in[4];
  const float* w3  = (const float*)d_in[5];
  const float* b3  = (const float*)d_in[6];
  const float* wl5 = (const float*)d_in[7];
  const float* bl5 = (const float*)d_in[8];
  const float* wl4 = (const float*)d_in[9];
  const float* bl4 = (const float*)d_in[10];
  const float* wl3 = (const float*)d_in[11];
  const float* bl3 = (const float*)d_in[12];
  const float* wf  = (const float*)d_in[13];
  const float* bfv = (const float*)d_in[14];
  float* outp = (float*)d_out;

  char* base = (char*)d_ws;
  size_t off = 0;
  auto alloc = [&](size_t bytes) -> char* {
    char* pp = base + off;
    off += (bytes + 255) & ~(size_t)255;
    return pp;
  };
  bf16* W1p = (bf16*)alloc((size_t)80 * 736 * 2);
  bf16* W3p = (bf16*)alloc((size_t)80 * 736 * 2);
  bf16* WL5 = (bf16*)alloc((size_t)576 * 1728 * 2);
  bf16* WL4 = (bf16*)alloc((size_t)576 * 1088 * 2);
  bf16* WL3 = (bf16*)alloc((size_t)576 * 640 * 2);
  bf16* WFp = (bf16*)alloc((size_t)64 * 576 * 2);
  float* b1p = (float*)alloc(80 * 4);
  float* b3p = (float*)alloc(80 * 4);
  float* xA  = (float*)alloc((size_t)BATCH * 64 * 36 * 4);
  float* xB  = (float*)alloc((size_t)BATCH * 64 * 36 * 4);
  bf16* Y1a  = (bf16*)alloc((size_t)4 * BATCH * 49 * 80 * 2);  // conv1 out, halo'd
  bf16* bufA = (bf16*)alloc((size_t)4 * BATCH * 49 * 80 * 2);  // Zwin / BM3 alias

  auto cdiv = [](long a, long b) { return (int)((a + b - 1) / b); };
  const int NB = 4 * BATCH;  // 32768

  pack_conv<<<cdiv((long)80 * 736, 256), 256, 0, stream>>>(w1, W1p);
  pack_conv<<<cdiv((long)80 * 736, 256), 256, 0, stream>>>(w3, W3p);
  pack_mat<<<cdiv((long)576 * 1728, 256), 256, 0, stream>>>(wl5, WL5, 576, 1700, 1728, 576);
  pack_mat<<<cdiv((long)576 * 1088, 256), 256, 0, stream>>>(wl4, WL4, 576, 1088, 1088, 576);
  pack_mat<<<cdiv((long)576 * 640, 256), 256, 0, stream>>>(wl3, WL3, 576, 612, 640, 576);
  pack_mat<<<cdiv((long)64 * 576, 256), 256, 0, stream>>>(wf, WFp, 64, 576, 576, 64);
  pack_vecf<<<1, 256, 0, stream>>>(b1, b1p, 68, 80);
  pack_vecf<<<1, 256, 0, stream>>>(b3, b3p, 68, 80);

  struct StageCfg { int ws; const float* xsrc; float* xdst; const bf16* WL;
                    const float* bl; int Ktl; int KlPad; int Kreal; };
  StageCfg stages[3] = {
    {5, x_in, xA, WL5, bl5, 54, 1728, 1700},
    {4, xA,   xB, WL4, bl4, 34, 1088, 1088},
    {3, xB,   xA, WL3, bl3, 20, 640,  612},
  };

  for (int s = 0; s < 3; ++s) {
    const StageCfg& c = stages[s];
    int ws = c.ws, ws2 = ws * ws;
    int wsp2 = (ws + 2) * (ws + 2);
    int N = NB * ws2;
    long totalZ = (long)NB * wsp2 * 80;
    act_win<<<cdiv(totalZ, 256), 256, 0, stream>>>(c.xsrc, cellp, iscp, bufA, ws);
    hipMemsetAsync(Y1a, 0, (size_t)NB * wsp2 * 80 * 2, stream);  // zero halo
    gemm_k<0, 5><<<dim3(N / 128, 1), 128, 0, stream>>>(W1p, b1p, bufA, Y1a,
        nullptr, nullptr, nullptr, 23, 736, ws, 0);
    gemm_k<1, 5><<<dim3(N / 128, 1), 128, 0, stream>>>(W3p, b3p, Y1a, bufA,
        nullptr, nullptr, nullptr, 23, 736, ws, c.KlPad);
    if (c.Kreal < c.KlPad)
      zero_pad_rows<<<cdiv((long)NB * (c.KlPad - c.Kreal), 256), 256, 0, stream>>>(
          bufA, c.Kreal, c.KlPad, NB);
    gemm_k<2, 4><<<dim3(NB / 128, 9), 128, 0, stream>>>(c.WL, c.bl, bufA,
        nullptr, c.xdst, nullptr, nullptr, c.Ktl, c.KlPad, ws, c.KlPad);
  }

  gemm_k<3, 4><<<dim3(NB / 128, 1), 128, 0, stream>>>(WFp, bfv, nullptr,
      nullptr, nullptr, outp, xA, 18, 576, 3, 0);
}
